// GRUGAN_29764123361548
// MI455X (gfx1250) — matrix operand order repaired and checked by reading, not compiled, not run on hardware
//
#include <hip/hip_runtime.h>
#include <cstddef>

typedef __attribute__((ext_vector_type(16))) _Float16 v16h;
typedef __attribute__((ext_vector_type(8)))  _Float16 v8h;
typedef __attribute__((ext_vector_type(8)))  float    v8f;
typedef __attribute__((ext_vector_type(4)))  float    v4f;

#define TT 512   // sequence length
#define BB 512   // batch
#define DD 32    // input dim
#define HH 64    // hidden dim
#define GN 192   // 3*H gate columns

// ---- fast transcendentals (gfx1250 TRANS ops where available) --------------
__device__ __forceinline__ float exp2_fast(float v) {
#if __has_builtin(__builtin_amdgcn_exp2f)
  return __builtin_amdgcn_exp2f(v);
#else
  return __expf(v * 0.6931471805599453f);
#endif
}
__device__ __forceinline__ float rcp_fast(float v) {
#if __has_builtin(__builtin_amdgcn_rcpf)
  return __builtin_amdgcn_rcpf(v);
#else
  return 1.0f / v;
#endif
}
__device__ __forceinline__ float sigmoid_fast(float v) {
  return rcp_fast(1.0f + exp2_fast(-1.4426950408889634f * v));
}
__device__ __forceinline__ float tanh_fast(float v) {
#if __has_builtin(__builtin_amdgcn_tanhf)
  return __builtin_amdgcn_tanhf(v);
#else
  return 2.0f * sigmoid_fast(2.0f * v) - 1.0f;
#endif
}

// ---------------------------------------------------------------------------
// Forward GRU recurrence. One workgroup = 16 batch rows, persistent over all
// 512 timesteps. Gate-symmetric column split: wave w owns the SAME 16-column
// slice of each gate (r: w*16.., z: 64+w*16.., n: 128+w*16..). Therefore each
// lane's WMMA accumulators hold r, z, n-x, n-h for the SAME (row, col) — the
// entire GRU cell update runs in-register on the accumulators, with 4 bias
// scalars per lane. No gate LDS traffic at all.
//
// fp32 master h lives column-major in the accumulator lanes' registers
// (lane -> rows lhalf*8+v, col wcol). The only LDS traffic is the f16 shadow
// of h (row-major, WMMA-A layout) needed by next step's A1/A2 — an inherent
// lane transpose — ping-ponged so a SINGLE barrier per step suffices.
//
// The combined [W_ih^T ; W_hh^T] (K=96) lives in VGPRs as 9 WMMA B-fragments
// for the whole loop; x_{t+1} is register-prefetched during step t.
// ---------------------------------------------------------------------------
__global__ __launch_bounds__(128) void gru_fwd_wmma(
    const float* __restrict__ x, const float* __restrict__ w_ih,
    const float* __restrict__ w_hh, const float* __restrict__ b_ih,
    const float* __restrict__ b_hh, float* __restrict__ hF)
{
  __shared__ __align__(16) _Float16 hbuf[2][16 * HH];   // ping-pong f16 h shadow

  const int tid   = threadIdx.x;
  const int uwave = __builtin_amdgcn_readfirstlane(tid >> 5);  // SGPR wave id
  const int lane  = tid & 31;
  const int lhalf = lane >> 4;   // 0/1: half-wave
  const int lmod  = lane & 15;
  const int b0    = blockIdx.x * 16;
  const int wcol  = uwave * 16 + lmod;   // this wave's column within each gate

  for (int i = tid; i < 16 * HH; i += 128) hbuf[0][i] = (_Float16)0.0f;

  // B fragments (weights) resident in VGPRs for the whole time loop.
  // B layout (32x16 f16): lane -> col n = lmod; element i -> k = (i/8)*16 + lhalf*8 + i%8.
  // Gate-symmetric columns: tile nt covers gate nt, columns nt*64 + uwave*16 ...
  v16h Bf[3][3];
  #pragma unroll
  for (int nt = 0; nt < 3; ++nt) {
    const int n = nt * 64 + uwave * 16 + lmod;
    #pragma unroll
    for (int ks = 0; ks < 3; ++ks) {
      v16h f;
      #pragma unroll
      for (int i = 0; i < 16; ++i) {
        const int kg = ks * 32 + (i >> 3) * 16 + lhalf * 8 + (i & 7);   // 0..95 combined K
        const float w = (kg < DD) ? w_ih[n * DD + kg]
                                  : w_hh[n * HH + (kg - DD)];
        f[i] = (_Float16)w;
      }
      Bf[ks][nt] = f;
    }
  }

  // Per-lane cell-update constants: one gate column each.
  const float br  = b_ih[wcol]       + b_hh[wcol];
  const float bz  = b_ih[64 + wcol]  + b_hh[64 + wcol];
  const float bnx = b_ih[128 + wcol];
  const float bnh = b_hh[128 + wcol];

  // fp32 master h, column-major in registers: hcol[v] = h[lhalf*8+v][wcol].
  float hcol[8];
  #pragma unroll
  for (int v = 0; v < 8; ++v) hcol[v] = 0.0f;

  // A layout (16x32 f16): lane -> row m = lmod; element i -> k = (i/8)*16 + lhalf*8 + i%8.
  const float* xrow = x + (size_t)(b0 + lmod) * TT * DD;
  const int kb = lhalf * 8;

  v4f xp[4];
  xp[0] = *(const v4f*)(xrow + kb);
  xp[1] = *(const v4f*)(xrow + kb + 4);
  xp[2] = *(const v4f*)(xrow + 16 + kb);
  xp[3] = *(const v4f*)(xrow + 16 + kb + 4);

  __syncthreads();

  for (int t = 0; t < TT; ++t) {
    // A0: x_t fragment, from prefetched registers.
    v16h A0;
    #pragma unroll
    for (int i = 0; i < 16; ++i) A0[i] = (_Float16)xp[i >> 2][i & 3];

    // Prefetch x_{t+1} while the WMMAs run (uniform select, no divergence).
    const int tn = (t + 1 < TT) ? (t + 1) : t;
    xp[0] = *(const v4f*)(xrow + tn * DD + kb);
    xp[1] = *(const v4f*)(xrow + tn * DD + kb + 4);
    xp[2] = *(const v4f*)(xrow + tn * DD + 16 + kb);
    xp[3] = *(const v4f*)(xrow + tn * DD + 16 + kb + 4);

    // A1/A2: h fragments straight from the current f16 shadow — no converts.
    const _Float16* hr = hbuf[t & 1] + lmod * HH + kb;
    v8h a1l = *(const v8h*)(hr);
    v8h a1h = *(const v8h*)(hr + 16);
    v8h a2l = *(const v8h*)(hr + 32);
    v8h a2h = *(const v8h*)(hr + 48);
    v16h A1 = __builtin_shufflevector(a1l, a1h, 0, 1, 2, 3, 4, 5, 6, 7,
                                      8, 9, 10, 11, 12, 13, 14, 15);
    v16h A2 = __builtin_shufflevector(a2l, a2h, 0, 1, 2, 3, 4, 5, 6, 7,
                                      8, 9, 10, 11, 12, 13, 14, 15);

    // r and z: x+h folded by the matrix unit (chained accumulator).
    v8f ar = {};
    ar = __builtin_amdgcn_wmma_f32_16x16x32_f16(false, A0, false, Bf[0][0],
                                                (short)0, ar, false, false);
    ar = __builtin_amdgcn_wmma_f32_16x16x32_f16(false, A1, false, Bf[1][0],
                                                (short)0, ar, false, false);
    ar = __builtin_amdgcn_wmma_f32_16x16x32_f16(false, A2, false, Bf[2][0],
                                                (short)0, ar, false, false);
    v8f az = {};
    az = __builtin_amdgcn_wmma_f32_16x16x32_f16(false, A0, false, Bf[0][1],
                                                (short)0, az, false, false);
    az = __builtin_amdgcn_wmma_f32_16x16x32_f16(false, A1, false, Bf[1][1],
                                                (short)0, az, false, false);
    az = __builtin_amdgcn_wmma_f32_16x16x32_f16(false, A2, false, Bf[2][1],
                                                (short)0, az, false, false);
    // n: x-part and h-part separate (b_hn sits inside r*(...)).
    v8f ax = {};
    v8f ah = {};
    ax = __builtin_amdgcn_wmma_f32_16x16x32_f16(false, A0, false, Bf[0][2],
                                                (short)0, ax, false, false);
    ah = __builtin_amdgcn_wmma_f32_16x16x32_f16(false, A1, false, Bf[1][2],
                                                (short)0, ah, false, false);
    ah = __builtin_amdgcn_wmma_f32_16x16x32_f16(false, A2, false, Bf[2][2],
                                                (short)0, ah, false, false);

    // GRU cell update directly on the accumulators (lane owns (rows, wcol)).
    _Float16* hw = hbuf[(t + 1) & 1];
    #pragma unroll
    for (int v = 0; v < 8; ++v) {
      const float r  = sigmoid_fast(ar[v] + br);
      const float z  = sigmoid_fast(az[v] + bz);
      const float nn = tanh_fast(ax[v] + bnx + r * (ah[v] + bnh));
      hcol[v] = (1.0f - z) * nn + z * hcol[v];
      hw[(lhalf * 8 + v) * HH + wcol] = (_Float16)hcol[v];  // f16 shadow (transpose)
    }

    __syncthreads();   // single barrier: shadow(t+1) published before next reads
  }

  // Final hidden state straight from registers.
  #pragma unroll
  for (int v = 0; v < 8; ++v)
    hF[(size_t)(b0 + lhalf * 8 + v) * HH + wcol] = hcol[v];
}

// ---------------------------------------------------------------------------
// Backward direction: hs_b[0] is the state after processing ONLY t = T-1 from
// h0 = 0, so all h@W_hh terms vanish -> one tiny projection + elementwise.
// ---------------------------------------------------------------------------
__global__ __launch_bounds__(256) void gru_bwd_step(
    const float* __restrict__ x, const float* __restrict__ w_ih,
    const float* __restrict__ b_ih, const float* __restrict__ b_hh,
    float* __restrict__ hB)
{
  __shared__ float xs[4][DD];
  const int tid  = threadIdx.x;
  const int bblk = blockIdx.x * 4;

  if (tid < 4 * DD)
    xs[tid / DD][tid % DD] =
        x[(size_t)(bblk + tid / DD) * TT * DD + (TT - 1) * DD + (tid % DD)];
  __syncthreads();

  const int bl = tid >> 6;       // 0..3 batch rows per block
  const int j  = tid & 63;       // hidden column
  const float* xv = xs[bl];

  float ar = b_ih[j], az = b_ih[64 + j], an = b_ih[128 + j];
  #pragma unroll
  for (int k = 0; k < DD; ++k) {
    const float xk = xv[k];
    ar += xk * w_ih[(j)       * DD + k];
    az += xk * w_ih[(64 + j)  * DD + k];
    an += xk * w_ih[(128 + j) * DD + k];
  }
  const float r = sigmoid_fast(ar + b_hh[j]);
  const float z = sigmoid_fast(az + b_hh[64 + j]);
  const float n = tanh_fast(an + r * b_hh[128 + j]);
  hB[(size_t)(bblk + bl) * HH + j] = (1.0f - z) * n;   // + z*0
}

// ---------------------------------------------------------------------------
// Head: out[b] = fc2( leaky_relu_0.2( fc1([hF[b] | hB[b]]) ) )
// ---------------------------------------------------------------------------
__global__ __launch_bounds__(64) void head_kernel(
    const float* __restrict__ hF, const float* __restrict__ hB,
    const float* __restrict__ fc1w, const float* __restrict__ fc1b,
    const float* __restrict__ fc2w, const float* __restrict__ fc2b,
    float* __restrict__ out)
{
  __shared__ float last[2 * HH];
  __shared__ float red[64];
  const int b = blockIdx.x;
  const int j = threadIdx.x;

  last[j]      = hF[(size_t)b * HH + j];
  last[HH + j] = hB[(size_t)b * HH + j];
  __syncthreads();

  float a = fc1b[j];
  #pragma unroll 8
  for (int k = 0; k < 2 * HH; ++k) a += fc1w[j * (2 * HH) + k] * last[k];
  a = (a >= 0.0f) ? a : 0.2f * a;
  red[j] = a * fc2w[j];
  __syncthreads();

  if (j < 32) red[j] += red[j + 32];
  __syncthreads();
  if (j == 0) {
    float s = fc2b[0];
    #pragma unroll
    for (int k = 0; k < 32; ++k) s += red[k];
    out[b] = s;
  }
}

extern "C" void kernel_launch(void* const* d_in, const int* in_sizes, int n_in,
                              void* d_out, int out_size, void* d_ws, size_t ws_size,
                              hipStream_t stream) {
  (void)in_sizes; (void)n_in; (void)out_size; (void)ws_size;
  const float* x      = (const float*)d_in[0];
  const float* w_ih_f = (const float*)d_in[1];
  const float* w_hh_f = (const float*)d_in[2];
  const float* b_ih_f = (const float*)d_in[3];
  const float* b_hh_f = (const float*)d_in[4];
  const float* w_ih_b = (const float*)d_in[5];
  // w_hh_b (d_in[6]) is dead: hs_b[0] starts from h0 == 0.
  const float* b_ih_b = (const float*)d_in[7];
  const float* b_hh_b = (const float*)d_in[8];
  const float* fc1_w  = (const float*)d_in[9];
  const float* fc1_b  = (const float*)d_in[10];
  const float* fc2_w  = (const float*)d_in[11];
  const float* fc2_b  = (const float*)d_in[12];

  float* hF = (float*)d_ws;              // [512, 64]
  float* hB = hF + (size_t)BB * HH;      // [512, 64]

  gru_fwd_wmma<<<BB / 16, 128, 0, stream>>>(x, w_ih_f, w_hh_f, b_ih_f, b_hh_f, hF);
  gru_bwd_step<<<BB / 4, 256, 0, stream>>>(x, w_ih_b, b_ih_b, b_hh_b, hB);
  head_kernel<<<BB, 64, 0, stream>>>(hF, hB, fc1_w, fc1_b, fc2_w, fc2_b, (float*)d_out);
}
